// BottleneckAttention_54305566491357
// MI455X (gfx1250) — matrix operand order repaired and checked by reading, not compiled, not run on hardware
//
#include <hip/hip_runtime.h>
#include <hip/hip_bf16.h>
#include <stdint.h>

// ---------------------------------------------------------------------------
// Sizes (fixed by the reference)
// ---------------------------------------------------------------------------
#define B_ 2
#define C_ 512
#define M_ 16
#define T_ 128
#define H_ 8
#define D_ 64
#define S_ 2048          // M_*T_

typedef _Float16 h16;
typedef __attribute__((ext_vector_type(16))) _Float16 v16h;
typedef __attribute__((ext_vector_type(8)))  float    v8f;
typedef __attribute__((ext_vector_type(4)))  float    v4f;
typedef __attribute__((ext_vector_type(4)))  uint32_t v4u;

// ---------------------------------------------------------------------------
// WMMA helpers (CDNA5 gfx1250, wave32).
// A 16x32 f16 fragment: lane = row (lane&15); K chunks: lanes 0-15 hold
// K = [0..7]+[16..23], lanes 16-31 hold K = [8..15]+[24..31]  (ISA 7.12.2)
// B 32x16 f16 fragment: lane = col (lane&15); lanes 0-15 hold K=[0..7]+[16..23],
// lanes 16-31 hold K=[8..15]+[24..31] (two 16B loads, same order as A).
// C/D 16x16 f32: VGPR r, lane l -> row = r + 8*(l>>4), col = l&15.
// ---------------------------------------------------------------------------
__device__ __forceinline__ v8f wmma32(v16h a, v16h b, v8f c) {
  return __builtin_amdgcn_wmma_f32_16x16x32_f16(false, a, false, b,
                                                (short)0, c, false, false);
}

__device__ __forceinline__ v16h load_a_frag(const h16* p, int ld, int k0, int lane) {
  int row = lane & 15, kb = (lane >> 4) * 8;
  const h16* q = p + (size_t)row * ld + k0 + kb;
  union { v4u v[2]; v16h h; } u;
  u.v[0] = *(const v4u*)(q);
  u.v[1] = *(const v4u*)(q + 16);
  return u.h;
}

__device__ __forceinline__ v16h load_b_frag(const h16* p, int ld, int k0, int lane) {
  int n = lane & 15, kb = (lane >> 4) * 8;
  const h16* q = p + (size_t)n * ld + k0 + kb;
  union { v4u v[2]; v16h h; } u;
  u.v[0] = *(const v4u*)(q);
  u.v[1] = *(const v4u*)(q + 16);
  return u.h;
}

// ---------------------------------------------------------------------------
// 1) GroupNorm: one block per (b, group); 4 channels x 2048 spatial = 8192 el.
// ---------------------------------------------------------------------------
__global__ void gn_kernel(const float* __restrict__ x, const float* __restrict__ sc,
                          const float* __restrict__ bi, float* __restrict__ xn) {
  int bg = blockIdx.x;
  int b = bg >> 7, g = bg & 127;
  size_t base = ((size_t)b * C_ + g * 4) * S_;
  int tid = threadIdx.x;
  float s1 = 0.f, s2 = 0.f;
  for (int i = tid; i < 4 * S_; i += 256) {
    float v = x[base + i];
    s1 += v; s2 += v * v;
  }
#pragma unroll
  for (int off = 16; off; off >>= 1) {
    s1 += __shfl_xor(s1, off, 32);
    s2 += __shfl_xor(s2, off, 32);
  }
  __shared__ float red[18];
  int wave = tid >> 5, lane = tid & 31;
  if (lane == 0) { red[wave] = s1; red[8 + wave] = s2; }
  __syncthreads();
  if (tid == 0) {
    float a = 0.f, q = 0.f;
    for (int w = 0; w < 8; ++w) { a += red[w]; q += red[8 + w]; }
    float mu = a * (1.f / (4.f * S_));
    float var = q * (1.f / (4.f * S_)) - mu * mu;
    red[16] = mu;
    red[17] = rsqrtf(var + 1e-5f);
  }
  __syncthreads();
  float mu = red[16], rstd = red[17];
  for (int i = tid; i < 4 * S_; i += 256) {
    int c = g * 4 + (i >> 11);
    xn[base + i] = (x[base + i] - mu) * rstd * sc[c] + bi[c];
  }
}

// ---------------------------------------------------------------------------
// 2) Weight transpose + f16 convert: Wt[n][k] = W[k][n]
// ---------------------------------------------------------------------------
__global__ void wt_kernel(const float* __restrict__ w, h16* __restrict__ wt) {
  int idx = blockIdx.x * 256 + threadIdx.x;   // n*C + k
  int n = idx >> 9, kk = idx & 511;
  wt[idx] = (h16)w[kk * C_ + n];
}

// ---------------------------------------------------------------------------
// 3) 3x3 depthwise conv (SAME) for q/k/v in one pass; outputs f16 in the
//    GEMM A-layout (b, s, c) so WMMA A fragments are contiguous.
// ---------------------------------------------------------------------------
__global__ void dw_kernel(const float* __restrict__ xn,
                          const float* __restrict__ wq, const float* __restrict__ wk,
                          const float* __restrict__ wv,
                          h16* __restrict__ yq, h16* __restrict__ yk, h16* __restrict__ yv) {
  int idx = blockIdx.x * 256 + threadIdx.x;   // b*(C*S) + c*S + s
  int b = idx >> 20;
  int rem = idx & ((1 << 20) - 1);
  int c = rem >> 11;
  int s = rem & 2047;
  int m = s >> 7, t = s & 127;
  const float* xb = xn + ((size_t)b * C_ + c) * S_;
  float aq = 0.f, ak = 0.f, av = 0.f;
#pragma unroll
  for (int dm = 0; dm < 3; ++dm) {
    int mm = m + dm - 1;
#pragma unroll
    for (int dt = 0; dt < 3; ++dt) {
      int tt = t + dt - 1;
      float v = (mm >= 0 && mm < M_ && tt >= 0 && tt < T_) ? xb[mm * T_ + tt] : 0.f;
      int wi = (dm * 3 + dt) * C_ + c;
      aq += v * wq[wi]; ak += v * wk[wi]; av += v * wv[wi];
    }
  }
  size_t o = ((size_t)b * S_ + s) * C_ + c;
  yq[o] = (h16)aq; yk[o] = (h16)ak; yv[o] = (h16)av;
}

// ---------------------------------------------------------------------------
// 4) WMMA GEMM: O[b] = A[b] (S x C, f16, ld=C) @ Wt^T (C x C) [+ bias]
//    block = 256 threads = 8 waves; block tile 128(rows) x 64(cols);
//    each wave owns a 32x32 tile (2x2 WMMA tiles), K loop = 16 x 32.
//    MODE 0: f16 out, (b, s, C) row-major (scattered b16 stores)
//    MODE 1: f16 out, transposed (b, C, S)  (packed b128 stores)  -> for V
//    MODE 2: f32 out, transposed (b, C, S), + bias + residual     -> final
// ---------------------------------------------------------------------------
template <int MODE>
__global__ void gemm_kernel(const h16* __restrict__ A, const h16* __restrict__ Wt,
                            const float* __restrict__ bias, const float* __restrict__ resid,
                            void* __restrict__ outp) {
  int tid = threadIdx.x, lane = tid & 31, wave = tid >> 5;
  int b = blockIdx.z;
  int r0 = blockIdx.x * 128 + (wave >> 1) * 32;
  int c0 = blockIdx.y * 64 + (wave & 1) * 32;
  const h16* Ab = A + (size_t)b * S_ * C_;

  v8f acc[2][2] = {};
  for (int k0 = 0; k0 < C_; k0 += 32) {
    v16h a0 = load_a_frag(Ab + (size_t)r0 * C_, C_, k0, lane);
    v16h a1 = load_a_frag(Ab + (size_t)(r0 + 16) * C_, C_, k0, lane);
    v16h b0 = load_b_frag(Wt + (size_t)c0 * C_, C_, k0, lane);
    v16h b1 = load_b_frag(Wt + (size_t)(c0 + 16) * C_, C_, k0, lane);
    acc[0][0] = wmma32(a0, b0, acc[0][0]);
    acc[0][1] = wmma32(a0, b1, acc[0][1]);
    acc[1][0] = wmma32(a1, b0, acc[1][0]);
    acc[1][1] = wmma32(a1, b1, acc[1][1]);
  }

  int col = lane & 15, hi = lane >> 4;
#pragma unroll
  for (int rt = 0; rt < 2; ++rt) {
#pragma unroll
    for (int ct = 0; ct < 2; ++ct) {
      int Ccol = c0 + ct * 16 + col;
      int Rb = r0 + rt * 16 + 8 * hi;          // 8 consecutive rows per lane
      v8f av = acc[rt][ct];
      if (MODE == 0) {
        float bb = (bias != nullptr) ? bias[Ccol] : 0.f;
        h16* o = (h16*)outp + ((size_t)b * S_ + Rb) * C_ + Ccol;
#pragma unroll
        for (int r = 0; r < 8; ++r) o[(size_t)r * C_] = (h16)(av[r] + bb);
      } else if (MODE == 1) {
        union { h16 hv[8]; v4u v; } pk;
#pragma unroll
        for (int r = 0; r < 8; ++r) pk.hv[r] = (h16)av[r];
        *(v4u*)((h16*)outp + ((size_t)b * C_ + Ccol) * S_ + Rb) = pk.v;
      } else {
        float bb = bias[Ccol];
        size_t off = ((size_t)b * C_ + Ccol) * S_ + Rb;
        v4f x0 = *(const v4f*)(resid + off);
        v4f x1 = *(const v4f*)(resid + off + 4);
#pragma unroll
        for (int r = 0; r < 4; ++r) { x0[r] += av[r] + bb; x1[r] += av[r + 4] + bb; }
        *(v4f*)((float*)outp + off) = x0;
        *(v4f*)((float*)outp + off + 4) = x1;
      }
    }
  }
}

// ---------------------------------------------------------------------------
// 5) RoPE-2D, in place on (b, s, c) f16 q/k. Pairs (d, d+16) within each
//    32-wide half of a head; first half rotates by m, second by t.
// ---------------------------------------------------------------------------
__global__ void rope_kernel(h16* __restrict__ p) {
  int idx = blockIdx.x * 256 + threadIdx.x;   // [b:1][s:11][h:3][i:5]
  int i = idx & 31;
  int h = (idx >> 5) & 7;
  int s = (idx >> 8) & 2047;
  int b = idx >> 19;
  size_t base = ((size_t)b * S_ + s) * C_ + h * 64;
  int j = i & 15;
  float pos = (i < 16) ? (float)(s >> 7) : (float)(s & 127);
  int p1 = (i < 16) ? j : 32 + j;
  int p2 = p1 + 16;
  float ang = pos * __powf(10000.f, -(float)j * (1.f / 16.f));
  float cc = __cosf(ang), sn = __sinf(ang);
  float v1 = (float)p[base + p1], v2 = (float)p[base + p2];
  p[base + p1] = (h16)(v1 * cc - v2 * sn);
  p[base + p2] = (h16)(v1 * sn + v2 * cc);
}

// ---------------------------------------------------------------------------
// 6) Attention: one block per (b, h, 16 query rows). Full 2048-key score row
//    lives in LDS as f16 (64 KB of the 320 KB WGP LDS), so softmax is a
//    simple two-pass with wave32 shuffle butterflies — no online rescaling.
//    Phase 1: 8 waves x 16 key-tiles each, 2 WMMAs (K=d, 64) per tile.
//    Phase 3: waves 0..3 each own 16 d-columns; PV with A from LDS (ds_load)
//    and B from transposed V (keys contiguous).
// ---------------------------------------------------------------------------
__global__ void attn_kernel(const h16* __restrict__ q, const h16* __restrict__ k,
                            const h16* __restrict__ vt, const int* __restrict__ lengths,
                            h16* __restrict__ o) {
  extern __shared__ char smem[];
  h16* sc = (h16*)smem;                         // [16][2048] f16 scores -> P
  float* rowsum = (float*)(smem + 16 * S_ * 2); // [16]

  int qt = blockIdx.x, h = blockIdx.y, b = blockIdx.z;
  int q0 = qt * 16;
  int tid = threadIdx.x, lane = tid & 31, wave = tid >> 5;
  int len = lengths[b];
  int col = lane & 15, hi = lane >> 4;

  const h16* qb = q + ((size_t)b * S_ + q0) * C_ + h * 64;
  const h16* kb = k + (size_t)b * S_ * C_ + h * 64;

  // Hoisted Q fragments (rows = queries, K = d in [0,32) and [32,64))
  v16h aq0 = load_a_frag(qb, C_, 0, lane);
  v16h aq1 = load_a_frag(qb, C_, 32, lane);

  // ---- Phase 1: scores -> LDS (scaled + key-masked) ----
  for (int it = 0; it < 16; ++it) {
    int kt = wave + 8 * it;                     // key tile 0..127
    const h16* kp = kb + (size_t)(kt * 16) * C_;
    v16h b0 = load_b_frag(kp, C_, 0, lane);     // B: n = key, K = d
    v16h b1 = load_b_frag(kp, C_, 32, lane);
    v8f c = {};
    c = wmma32(aq0, b0, c);
    c = wmma32(aq1, b1, c);
    int key = kt * 16 + col;
    bool valid = (key & (T_ - 1)) < len;
    float msc = valid ? 0.125f : 0.f;           // 1/sqrt(D)
    float mad = valid ? 0.f : -30000.f;         // f16-safe -inf
#pragma unroll
    for (int r = 0; r < 8; ++r)
      sc[(size_t)(r + 8 * hi) * S_ + key] = (h16)(c[r] * msc + mad);
  }
  __syncthreads();

  // ---- Phase 2: row softmax (16 threads per row, shuffle reduce) ----
  {
    int row = tid >> 4, sub = tid & 15;
    float mx = -1e30f;
    for (int i = sub; i < S_; i += 16) mx = fmaxf(mx, (float)sc[(size_t)row * S_ + i]);
#pragma unroll
    for (int off = 8; off; off >>= 1) mx = fmaxf(mx, __shfl_xor(mx, off, 16));
    float sm = 0.f;
    for (int i = sub; i < S_; i += 16) {
      float p = __expf((float)sc[(size_t)row * S_ + i] - mx);
      sc[(size_t)row * S_ + i] = (h16)p;        // in-place: score -> P
      sm += p;
    }
#pragma unroll
    for (int off = 8; off; off >>= 1) sm += __shfl_xor(sm, off, 16);
    if (sub == 0) rowsum[row] = sm;
  }
  __syncthreads();

  // ---- Phase 3: O = P @ V (waves 0..3, one 16-wide d tile each) ----
  if (wave < 4) {
    int d0 = wave * 16;
    const h16* vb = vt + ((size_t)b * C_ + h * 64 + d0) * S_;  // rows = d, ld = S
    v8f acc = {};
    for (int kt2 = 0; kt2 < 64; ++kt2) {
      int k0 = kt2 * 32;
      int rowa = lane & 15, kba = (lane >> 4) * 8;
      union { v4u v[2]; v16h hh; } ua;          // A fragment from LDS
      ua.v[0] = *(const v4u*)&sc[(size_t)rowa * S_ + k0 + kba];
      ua.v[1] = *(const v4u*)&sc[(size_t)rowa * S_ + k0 + 16 + kba];
      v16h bf = load_b_frag(vb, S_, k0, lane);  // B: n = d, K = keys
      acc = wmma32(ua.hh, bf, acc);
    }
    h16* ob = o + ((size_t)b * S_ + q0) * C_ + h * 64 + d0 + col;
#pragma unroll
    for (int r = 0; r < 8; ++r) {
      int row = r + 8 * hi;
      ob[(size_t)row * C_] = (h16)(acc[r] / rowsum[row]);
    }
  }
}

// ---------------------------------------------------------------------------
// Host launcher
// ---------------------------------------------------------------------------
extern "C" void kernel_launch(void* const* d_in, const int* in_sizes, int n_in,
                              void* d_out, int out_size, void* d_ws, size_t ws_size,
                              hipStream_t stream) {
  const float* x        = (const float*)d_in[0];
  const int*   lengths  = (const int*)d_in[1];
  const float* gn_scale = (const float*)d_in[2];
  const float* gn_bias  = (const float*)d_in[3];
  const float* dw_q     = (const float*)d_in[4];
  const float* dw_k     = (const float*)d_in[5];
  const float* dw_v     = (const float*)d_in[6];
  const float* pw_q     = (const float*)d_in[7];
  const float* pw_k     = (const float*)d_in[8];
  const float* pw_v     = (const float*)d_in[9];
  const float* attn_w   = (const float*)d_in[10];
  const float* attn_b   = (const float*)d_in[11];
  const float* out_w    = (const float*)d_in[12];
  const float* out_b    = (const float*)d_in[13];

  // Workspace carving (~42.5 MB total)
  char* w = (char*)d_ws;
  const size_t F32_ACT = (size_t)B_ * C_ * S_ * sizeof(float);  // 8 MB
  const size_t F16_ACT = (size_t)B_ * S_ * C_ * sizeof(h16);    // 4 MB
  const size_t F16_W   = (size_t)C_ * C_ * sizeof(h16);         // 512 KB
  float* xn  = (float*)w;  w += F32_ACT;
  h16* yq    = (h16*)w;    w += F16_ACT;
  h16* yk    = (h16*)w;    w += F16_ACT;
  h16* yv    = (h16*)w;    w += F16_ACT;
  h16* qb    = (h16*)w;    w += F16_ACT;
  h16* kbuf  = (h16*)w;    w += F16_ACT;
  h16* vtb   = (h16*)w;    w += F16_ACT;   // V transposed (b, C, S)
  h16* ob    = (h16*)w;    w += F16_ACT;   // attention out (b, s, C)
  h16* opb   = (h16*)w;    w += F16_ACT;   // after attn_w  (b, s, C)
  h16* wtq   = (h16*)w;    w += F16_W;
  h16* wtk   = (h16*)w;    w += F16_W;
  h16* wtv   = (h16*)w;    w += F16_W;
  h16* wta   = (h16*)w;    w += F16_W;
  h16* wto   = (h16*)w;    w += F16_W;

  // Weight transpose + f16 convert
  wt_kernel<<<C_ * C_ / 256, 256, 0, stream>>>(pw_q, wtq);
  wt_kernel<<<C_ * C_ / 256, 256, 0, stream>>>(pw_k, wtk);
  wt_kernel<<<C_ * C_ / 256, 256, 0, stream>>>(pw_v, wtv);
  wt_kernel<<<C_ * C_ / 256, 256, 0, stream>>>(attn_w, wta);
  wt_kernel<<<C_ * C_ / 256, 256, 0, stream>>>(out_w, wto);

  // GroupNorm
  gn_kernel<<<B_ * 128, 256, 0, stream>>>(x, gn_scale, gn_bias, xn);

  // Depthwise 3x3 (q,k,v fused)
  dw_kernel<<<(B_ * C_ * S_) / 256, 256, 0, stream>>>(xn, dw_q, dw_k, dw_v, yq, yk, yv);

  // Pointwise projections (WMMA GEMMs)
  dim3 gg(S_ / 128, C_ / 64, B_);
  gemm_kernel<0><<<gg, 256, 0, stream>>>(yq, wtq, nullptr, nullptr, qb);
  gemm_kernel<0><<<gg, 256, 0, stream>>>(yk, wtk, nullptr, nullptr, kbuf);
  gemm_kernel<1><<<gg, 256, 0, stream>>>(yv, wtv, nullptr, nullptr, vtb);

  // RoPE-2D on q and k
  rope_kernel<<<(B_ * S_ * H_ * 32) / 256, 256, 0, stream>>>(qb);
  rope_kernel<<<(B_ * S_ * H_ * 32) / 256, 256, 0, stream>>>(kbuf);

  // Attention (dynamic LDS: 16x2048 f16 scores + 16 f32 row sums)
  size_t lds = (size_t)16 * S_ * sizeof(h16) + 16 * sizeof(float);
  attn_kernel<<<dim3(S_ / 16, H_, B_), 256, lds, stream>>>(qb, kbuf, vtb, lengths, ob);

  // Output projections: attn_w (+bias) then out_w (+bias+residual, f32 out)
  gemm_kernel<0><<<gg, 256, 0, stream>>>(ob, wta, attn_b, nullptr, opb);
  gemm_kernel<2><<<gg, 256, 0, stream>>>(opb, wto, out_b, x, d_out);
}
